// DeltaGRU_53807350284309
// MI455X (gfx1250) — matrix operand order repaired and checked by reading, not compiled, not run on hardware
//
#include <hip/hip_runtime.h>
#include <cmath>

// ---------------------------------------------------------------------------
// DeltaGRU on gfx1250 (MI455X).
//   B=128, T=4096, H=512, N=3H=1536.
//   grid = 8 workgroups (16 batch rows each), block = 1024 threads = 32 waves.
//   wave w owns hidden slice j in [16w, 16w+16) -> WMMA N-tiles {w, 32+w, 64+w}.
//   K = 17 chunks: 0..15 = dh (H=512), 16 = dx (K=6 zero-padded), so the whole
//   x-path GEMM rides the WMMA stream. 51 x v_wmma_f32_16x16x32_bf16 per
//   wave per step; the GRU "dm" state lives in the persistent f32 accumulators.
// ---------------------------------------------------------------------------

typedef __attribute__((ext_vector_type(16))) __bf16 bf16x16;
typedef __attribute__((ext_vector_type(8)))  float  f32x8;

#define T_LEN    4096
#define H_DIM    512
#define NTILES   96          // 1536 / 16
#define KCHUNKS  16          // 512 / 32  (dh chunks)
#define KCHUNKT  17          // + 1 dx chunk
#define WAVES    32
#define THREADS  1024
#define TH_X     0.1f

__device__ __forceinline__ float hswish(float v) {
  float c = fminf(fmaxf(v + 3.f, 0.f), 6.f);
  return v * c * (1.f / 6.f);
}

// ---------------------------------------------------------------------------
// Pack B-fragments (bf16):
//   chunks 0..15 from w_h2h fp32 [n=1536][k=512]
//   chunk  16    from w_x2h fp32 [n=1536][k=6], zero padded to K=32
// B 32x16 bf16 (wave32), contiguous-k order chosen here (not the ISA table): lanes 0-15 hold N=lane&15, K=0..15 (V0..7);
// lanes 16-31 hold K=16..31.  Each lane's 16 values contiguous:
//   packed[((c*96 + t)*32 + lane)*16 + i] = B[k = c*32 + (lane>>4)*16 + i][n]
// ---------------------------------------------------------------------------
__global__ void pack_b(const float* __restrict__ wh, const float* __restrict__ wx,
                       __bf16* __restrict__ p) {
  int idx = blockIdx.x * 256 + threadIdx.x;
  if (idx >= KCHUNKT * NTILES * 32 * 16) return;
  int i    = idx & 15;
  int lane = (idx >> 4) & 31;
  int rest = idx >> 9;
  int t    = rest % NTILES;
  int c    = rest / NTILES;
  int n    = t * 16 + (lane & 15);
  int kloc = ((lane >> 4) * 16) + i;
  if (c < KCHUNKS) {
    p[idx] = (__bf16)wh[n * H_DIM + c * 32 + kloc];
  } else {
    p[idx] = (kloc < 6) ? (__bf16)wx[n * 6 + kloc] : (__bf16)0.f;
  }
}

__launch_bounds__(THREADS, 1)
__global__ void deltagru_main(const float* __restrict__ x,
                              const float* __restrict__ w_fc,
                              const float* __restrict__ w_conv1,
                              const float* __restrict__ w_conv2,
                              const __bf16* __restrict__ packedB,
                              float* __restrict__ out) {
  // LDS: A-fragment double buffer (17 chunks: dh + dx, bf16) + fc partials.
  __shared__ __bf16 Abuf[2 * KCHUNKT * 32 * 16];   // 34 KB
  __shared__ float  outbuf[WAVES][16][2];          // 4 KB

  const int tid    = threadIdx.x;
  const int lane   = tid & 31;
  const int wave   = tid >> 5;
  const int laneHi = lane >> 4;
  const int jl     = lane & 15;            // column within tile
  const int j      = wave * 16 + jl;       // owned hidden index (C column)
  const int bblk   = blockIdx.x * 16;      // batch base for this WG

  const float wfc0 = w_fc[j];
  const float wfc1 = w_fc[512 + j];

  // Persistent state (fp32 in VGPRs). All four are WMMA C/D accumulators:
  //   accR = dm_r, accZ = dm_z, accN = dm_n (x-parts), accNH = dm_nh (h-parts)
  f32x8 accR = {}, accZ = {}, accNH = {}, accN = {};
  float h[8] = {};
  float xp0 = 0.f, xp1 = 0.f, xp2 = 0.f;   // wave0 delta-x state (3 feats/lane)

  // Loop-invariant dx-chunk B fragments: keep resident for all 4096 steps.
  const bf16x16 bxR = *(const bf16x16*)&packedB[((size_t)(KCHUNKS * NTILES +      wave) * 32 + lane) * 16];
  const bf16x16 bxZ = *(const bf16x16*)&packedB[((size_t)(KCHUNKS * NTILES + 32 + wave) * 32 + lane) * 16];
  const bf16x16 bxN = *(const bf16x16*)&packedB[((size_t)(KCHUNKS * NTILES + 64 + wave) * 32 + lane) * 16];

  // A-fragment (16-bit A 16x32, contiguous k = 16h+s as in pack_b) address pieces, fixed per lane:
  // k_local = j&31; laneA_hi = (k_local>>4)&1; slot = k_local&15
  const int klocal   = j & 31;
  const int chunkOfJ = j >> 5;
  const int laneAhi  = (klocal >> 4) & 1;
  const int slotA    = klocal & 15;

  // Zero A buffers (dh_0 = 0; dx pad region stays 0 forever).
  for (int q = tid; q < 2 * KCHUNKT * 32 * 16; q += THREADS) Abuf[q] = (__bf16)0.f;
  __syncthreads();

  int parity = 0;
  for (int step = 0; step < T_LEN; ++step) {
    // ---- phase A (parallel across waves 0/1) ------------------------------
    if (wave == 0) {
      // dx features -> A-fragment chunk 16 of the CURRENT parity buffer.
      // lane handles batch b = jl, features f = 3*laneHi .. 3*laneHi+2.
      const int b = jl;
      const size_t xb = (size_t)(bblk + b) * T_LEN * 2;
      const float xi = x[xb + (size_t)step * 2 + 0];
      const float xq = x[xb + (size_t)step * 2 + 1];
      const int   tn = (step + 1 == T_LEN) ? 0 : step + 1;
      const float amp = sqrtf(xi * xi + xq * xq);
      float f0, f1, f2;
      if (laneHi == 0) { f0 = xi; f1 = xq; f2 = amp; }
      else {
        f0 = amp * amp * amp;
        f1 = x[xb + (size_t)tn * 2 + 0];
        f2 = x[xb + (size_t)tn * 2 + 1];
      }
      float d0 = f0 - xp0, a0 = fabsf(d0);
      float d1 = f1 - xp1, a1 = fabsf(d1);
      float d2 = f2 - xp2, a2 = fabsf(d2);
      // A chunk-16 layout: k_local = f in [0,6) -> laneA = m = b, slot = f.
      __bf16* Adx = &Abuf[(parity * KCHUNKT + KCHUNKS) * (32 * 16) + b * 16 + laneHi * 3];
      Adx[0] = (__bf16)((a0 < TH_X) ? 0.f : d0);
      Adx[1] = (__bf16)((a1 < TH_X) ? 0.f : d1);
      Adx[2] = (__bf16)((a2 < TH_X) ? 0.f : d2);
      xp0 = (a0 >= TH_X) ? f0 : xp0;
      xp1 = (a1 >= TH_X) ? f1 : xp1;
      xp2 = (a2 >= TH_X) ? f2 : xp2;
    } else if (wave == 1 && step > 0) {
      // Output row for step-1: fc reduction result + conv skip path.
      const int b = jl, o = laneHi, tt = step - 1;
      float s = 0.f;
#pragma unroll
      for (int wv = 0; wv < WAVES; ++wv) s += outbuf[wv][b][o];
      const size_t xb = (size_t)(bblk + b) * T_LEN * 2;
      float yv[3];
#pragma unroll
      for (int oc = 0; oc < 3; ++oc) {
        float a = 0.f;
#pragma unroll
        for (int kk = 0; kk < 3; ++kk) {
          int ts = tt + 16 * (kk - 1);
          if (ts >= 0 && ts < T_LEN) {
            a += x[xb + (size_t)ts * 2 + 0] * w_conv1[oc * 6 + 0 * 3 + kk];
            a += x[xb + (size_t)ts * 2 + 1] * w_conv1[oc * 6 + 1 * 3 + kk];
          }
        }
        yv[oc] = hswish(a);
      }
      float zv = yv[0] * w_conv2[o * 3 + 0] + yv[1] * w_conv2[o * 3 + 1] +
                 yv[2] * w_conv2[o * 3 + 2];
      out[((size_t)(bblk + b) * T_LEN + tt) * 2 + o] = s + hswish(zv);
    }
    __syncthreads();

    // ---- phase C: full mac via WMMA ---------------------------------------
    {
      const __bf16* Ab = &Abuf[parity * KCHUNKT * 32 * 16];
#pragma unroll
      for (int c = 0; c < KCHUNKS; ++c) {
        bf16x16 a  = *(const bf16x16*)&Ab[(c * 32 + lane) * 16];
        bf16x16 br = *(const bf16x16*)&packedB[((size_t)(c * NTILES +      wave) * 32 + lane) * 16];
        bf16x16 bz = *(const bf16x16*)&packedB[((size_t)(c * NTILES + 32 + wave) * 32 + lane) * 16];
        bf16x16 bn = *(const bf16x16*)&packedB[((size_t)(c * NTILES + 64 + wave) * 32 + lane) * 16];
        accR  = __builtin_amdgcn_wmma_f32_16x16x32_bf16(false, a, false, br, (short)0, accR,  false, false);
        accZ  = __builtin_amdgcn_wmma_f32_16x16x32_bf16(false, a, false, bz, (short)0, accZ,  false, false);
        accNH = __builtin_amdgcn_wmma_f32_16x16x32_bf16(false, a, false, bn, (short)0, accNH, false, false);
      }
      // dx chunk (B fragments resident in registers); n-tile goes to dm_n.
      bf16x16 a16 = *(const bf16x16*)&Ab[(KCHUNKS * 32 + lane) * 16];
      accR = __builtin_amdgcn_wmma_f32_16x16x32_bf16(false, a16, false, bxR, (short)0, accR, false, false);
      accZ = __builtin_amdgcn_wmma_f32_16x16x32_bf16(false, a16, false, bxZ, (short)0, accZ, false, false);
      accN = __builtin_amdgcn_wmma_f32_16x16x32_bf16(false, a16, false, bxN, (short)0, accN, false, false);
    }

    // ---- phase D: gates, h update, dh -> A-fragments, fc partials ---------
    {
      float pv0[8], pv1[8];
      __bf16* Aw = &Abuf[((parity ^ 1) * KCHUNKT + chunkOfJ) * (32 * 16)];
#pragma unroll
      for (int v = 0; v < 8; ++v) {
        const float r  = 1.f / (1.f + __expf(-accR[v]));
        const float z  = 1.f / (1.f + __expf(-accZ[v]));
        const float n  = tanhf(accN[v] + r * accNH[v]);
        const float hn = (1.f - z) * n + z * h[v];
        const float dh = hn - h[v];
        h[v] = hn;
        const int m = v + 8 * laneHi;
        Aw[(m + 16 * laneAhi) * 16 + slotA] = (__bf16)dh;   // TH_H==0 -> dh always
        pv0[v] = hn * wfc0;
        pv1[v] = hn * wfc1;
      }
      // reduce over the 16 columns of each half-wave (lane bits 0..3)
#pragma unroll
      for (int mask = 1; mask < 16; mask <<= 1) {
#pragma unroll
        for (int v = 0; v < 8; ++v) {
          pv0[v] += __shfl_xor(pv0[v], mask, 32);
          pv1[v] += __shfl_xor(pv1[v], mask, 32);
        }
      }
      if (jl == 0) {
#pragma unroll
        for (int v = 0; v < 8; ++v) {
          outbuf[wave][v + 8 * laneHi][0] = pv0[v];
          outbuf[wave][v + 8 * laneHi][1] = pv1[v];
        }
      }
    }
    parity ^= 1;
    __syncthreads();
  }

  // Flush output for t = T-1.
  if (wave == 1) {
    const int b = jl, o = laneHi, tt = T_LEN - 1;
    float s = 0.f;
#pragma unroll
    for (int wv = 0; wv < WAVES; ++wv) s += outbuf[wv][b][o];
    const size_t xb = (size_t)(bblk + b) * T_LEN * 2;
    float yv[3];
#pragma unroll
    for (int oc = 0; oc < 3; ++oc) {
      float a = 0.f;
#pragma unroll
      for (int kk = 0; kk < 3; ++kk) {
        int ts = tt + 16 * (kk - 1);
        if (ts >= 0 && ts < T_LEN) {
          a += x[xb + (size_t)ts * 2 + 0] * w_conv1[oc * 6 + 0 * 3 + kk];
          a += x[xb + (size_t)ts * 2 + 1] * w_conv1[oc * 6 + 1 * 3 + kk];
        }
      }
      yv[oc] = hswish(a);
    }
    float zv = yv[0] * w_conv2[o * 3 + 0] + yv[1] * w_conv2[o * 3 + 1] +
               yv[2] * w_conv2[o * 3 + 2];
    out[((size_t)(bblk + b) * T_LEN + tt) * 2 + o] = s + hswish(zv);
  }
}

extern "C" void kernel_launch(void* const* d_in, const int* in_sizes, int n_in,
                              void* d_out, int out_size, void* d_ws, size_t ws_size,
                              hipStream_t stream) {
  (void)in_sizes; (void)n_in; (void)out_size; (void)ws_size;
  const float* x       = (const float*)d_in[0];
  // d_in[1] = h_0 (all zeros) -- reference carry starts at zero anyway
  const float* w_x2h   = (const float*)d_in[2];
  const float* w_h2h   = (const float*)d_in[3];
  const float* w_fc    = (const float*)d_in[4];
  const float* w_conv1 = (const float*)d_in[5];
  const float* w_conv2 = (const float*)d_in[6];
  __bf16* packedB = (__bf16*)d_ws;                 // ~1.63 MB of workspace

  const int packElems = KCHUNKT * NTILES * 32 * 16;  // 835584
  pack_b<<<(packElems + 255) / 256, 256, 0, stream>>>(w_h2h, w_x2h, packedB);

  deltagru_main<<<128 / 16, THREADS, 0, stream>>>(x, w_fc, w_conv1, w_conv2,
                                                  packedB, (float*)d_out);
}
